// D6BPixelMotifBranch_3719441678841
// MI455X (gfx1250) — compile-verified, matrix operand order checked by reading, not run on hardware
//
#include <hip/hip_runtime.h>
#include <hip/hip_bf16.h>

typedef __attribute__((ext_vector_type(16))) __bf16 v16bf;
typedef __attribute__((ext_vector_type(8)))  float  v8f;
typedef __attribute__((ext_vector_type(4)))  unsigned int u32x4;
typedef __attribute__((ext_vector_type(8)))  int i32x8;
typedef __attribute__((ext_vector_type(4)))  int i32x4;

// ---------------------------------------------------------------------------
// WMMA helpers (CDNA5 v_wmma_f32_16x16x32_bf16)
// ---------------------------------------------------------------------------
__device__ inline v8f wmma_bf(v16bf a, v16bf b, v8f c) {
  return __builtin_amdgcn_wmma_f32_16x16x32_bf16(false, a, false, b, (short)0, c,
                                                 false, false);
}

// Fragment layout (ISA 7.12.2): a 16x32 bf16 operand tile stored lane-major:
// 32 lanes x 16 halfs (32B per lane, 1KB per fragment, 32B aligned).
// Element (mn, kr):  lane = (mn&15) + 16*((kr>>3)&1),
//                    slot = (kr&7) + 8*((kr>>4)&1).
__device__ inline int frag_off(int mn, int kr) {
  return (((mn & 15) + (((kr >> 3) & 1) << 4)) << 4) +
         ((kr & 7) + (((kr >> 4) & 1) << 3));
}

__device__ inline v16bf ld_frag(const __bf16* frag, int lane) {
  return *(const v16bf*)(frag + (lane << 4));
}

__device__ inline float gelu_exact(float x) {
  return 0.5f * x * (1.0f + erff(x * 0.70710678118654752f));
}

// ---------------------------------------------------------------------------
// Tensor Data Mover: DMA a 64x64 f32 tile (contiguous rows, stride 64) from
// global into LDS. D# per ISA 8.3/8.4: group0 = {count=1, lds_addr,
// global_addr, type=2}; group1 = {data_size=4B, tensor_dim0/1=64,
// tile_dim0/1=64, tensor_dim0_stride=64}; remaining groups unused (<=2D).
// Issue from one wave only; tracked by that wave's TENSORcnt.
// NOTE: this toolchain exposes the 6-arg builtin
//   (u32x4, i32x8, i32x4, i32x4, i32x8, i32 cpol).
// ---------------------------------------------------------------------------
__device__ inline void tdm_load_tile(unsigned int lds_off, const float* gptr) {
  unsigned long long ga = (unsigned long long)(uintptr_t)gptr;
  u32x4 g0;
  g0[0] = 1u;                                   // count=1 (valid descriptor)
  g0[1] = lds_off;                              // lds_addr (bytes)
  g0[2] = (unsigned int)ga;                     // global_addr[31:0]
  g0[3] = (unsigned int)((ga >> 32) & 0x01FFFFFFu) | 0x80000000u;  // type=2
  i32x8 g1;
  g1[0] = 0x00020000;            // data_size=2 (4 bytes)
  g1[1] = (int)(64u << 16);      // tensor_dim0 = 64 (lo16 at bits 63:48)
  g1[2] = (int)(64u << 16);      // td0 hi=0 | tensor_dim1 = 64 (lo16)
  g1[3] = (int)(64u << 16);      // td1 hi=0 | tile_dim0 = 64
  g1[4] = 64;                    // tile_dim1 = 64, tile_dim2 = 0
  g1[5] = 64;                    // tensor_dim0_stride = 64 (lo32)
  g1[6] = 0;
  g1[7] = 0;
  i32x4 z4 = {0, 0, 0, 0};
  i32x8 z8 = {0, 0, 0, 0, 0, 0, 0, 0};
  __builtin_amdgcn_tensor_load_to_lds(g0, g1, z4, z4, z8, 0);
}

// ===========================================================================
// Stage 1: fused pixel pass (single read of the 151MB h_pixel tensor).
//   TDM double-buffers 64x64 f32 tiles into LDS; per tile:
//   kp = h@Wk+bk, vp = h@Wv+bv (WMMA), logits = kp @ qT (WMMA),
//   softmax over 16 slots, featext += m^T @ [vp | x | y | 1] (WMMA).
// Grid: (B, 3 segments of 768 pixels). Block: 256 threads (8 waves).
// ===========================================================================
__global__ __launch_bounds__(256) void pixel_stage(
    const float* __restrict__ h, const int* __restrict__ mask,
    const float* __restrict__ pq, const float* __restrict__ Wk,
    const float* __restrict__ bk, const float* __restrict__ Wv,
    const float* __restrict__ bv, float* __restrict__ d_part) {
  const int b = blockIdx.x, seg = blockIdx.y;
  const int tid = threadIdx.x, lane = tid & 31, wave = tid >> 5;

  __shared__ float sHraw[2][64 * 64];  // TDM ping-pong staging (f32)
  __shared__ v16bf sHF_v[8 * 32];      // A frags [rt(4)][kt(2)] pixels x feat
  __shared__ v16bf sWkF_v[8 * 32];     // B frags [kt(2)][ct(4)]
  __shared__ v16bf sWvF_v[8 * 32];     // B frags [kt(2)][ct(4)]
  __shared__ v16bf sQtF_v[2 * 32];     // B frags [kt(2)]        q^T 64x16
  __shared__ v16bf sKPF_v[8 * 32];     // A frags [rt(4)][kt(2)] kp
  __shared__ v16bf sVPF_v[10 * 32];    // B frags [kt(2)][ct(5)] vp-ext 64x80
  __shared__ v16bf sMF_v[2 * 32];      // A frags [kt(2)]        m^T 16x64
  __shared__ float sLog[64 * 17];
  __shared__ float sOut[16 * 80];
  __shared__ float sQn[16];
  __bf16* sHF  = (__bf16*)sHF_v;
  __bf16* sWkF = (__bf16*)sWkF_v;
  __bf16* sWvF = (__bf16*)sWvF_v;
  __bf16* sQtF = (__bf16*)sQtF_v;
  __bf16* sKPF = (__bf16*)sKPF_v;
  __bf16* sVPF = (__bf16*)sVPF_v;
  __bf16* sMF  = (__bf16*)sMF_v;

  const size_t hbase = ((size_t)b * 2304 + seg * 768) * 64;
  // kick off TDM for tile 0 (wave 0 owns TENSORcnt)
  if (wave == 0)
    tdm_load_tile((unsigned int)(uintptr_t)&sHraw[0][0], &h[hbase]);

  // --- stage weights into B-fragment layout (once, amortized) ---
  for (int i = tid; i < 64 * 64; i += 256) {
    int r = i >> 6, c = i & 63;  // r = K (input dim), c = output col
    int fo = ((r >> 5) * 4 + (c >> 4)) * 512 + frag_off(c, r & 31);
    sWkF[fo] = (__bf16)Wk[i];
    sWvF[fo] = (__bf16)Wv[i];
  }
  if (tid < 16) {
    float s = 0.0f;
    for (int d = 0; d < 64; ++d) { float v = pq[tid * 64 + d]; s += v * v; }
    sQn[tid] = fmaxf(sqrtf(s), 1e-6f);
  }
  // zero the ext-cols fragments of sVPF (cols 67..79 stay zero forever)
  for (int i = tid; i < 512; i += 256) {
    int kt = i >> 8, idx = (i & 255) << 1;
    *(uint32_t*)&sVPF[(kt * 5 + 4) * 512 + idx] = 0u;
  }
  __syncthreads();
  for (int i = tid; i < 16 * 64; i += 256) {
    int k = i >> 6, d = i & 63;  // k = slot col, d = K row
    sQtF[(d >> 5) * 512 + frag_off(k, d & 31)] = (__bf16)(pq[i] / sQn[k]);
  }

  v8f accP = {0, 0, 0, 0, 0, 0, 0, 0};
  const float inv47 = 1.0f / 47.0f;

  for (int it = 0; it < 12; ++it) {
    const int n0 = seg * 768 + it * 64;
    const int cur = it & 1;
    // wait for this tile's DMA, then barrier so all waves see the LDS data
    // (barrier also protects sVPF/sMF from the previous pool GEMM reads)
    if (wave == 0) __builtin_amdgcn_s_wait_tensorcnt(0);
    __syncthreads();
    // immediately issue next tile's DMA: overlaps with convert + 3 GEMMs
    if (wave == 0 && it + 1 < 12)
      tdm_load_tile((unsigned int)(uintptr_t)&sHraw[cur ^ 1][0],
                    &h[hbase + (size_t)(it + 1) * 64 * 64]);

    // staged f32 tile -> bf16 A-fragments (b128 LDS loads, 8B LDS stores)
    for (int i = tid; i < 1024; i += 256) {
      int r = i >> 4, c = (i & 15) << 2;
      const float4 v = *(const float4*)&sHraw[cur][(r << 6) + c];
      int kr = c & 31;
      __bf16* frag = &sHF[((r >> 4) * 2 + (c >> 5)) * 512];
      int lt = (r & 15) + (((kr >> 3) & 1) << 4);
      int slot = (kr & 7) + (((kr >> 4) & 1) << 3);
      union { __bf16 h4[4]; uint2 u; } pk;
      pk.h4[0] = (__bf16)v.x; pk.h4[1] = (__bf16)v.y;
      pk.h4[2] = (__bf16)v.z; pk.h4[3] = (__bf16)v.w;
      *(uint2*)&frag[lt * 16 + slot] = pk.u;
    }
    // ext columns of vp-ext: [x, y, 1] per pixel
    if (tid < 64) {
      int n = n0 + tid;
      float px = (float)(n % 48) * inv47;
      float py = (float)(n / 48) * inv47;
      int kr = tid & 31;
      __bf16* fr = &sVPF[((tid >> 5) * 5 + 4) * 512];
      int g16 = (((kr >> 3) & 1) << 4);
      int slot = (kr & 7) + (((kr >> 4) & 1) << 3);
      fr[(0 + g16) * 16 + slot] = (__bf16)px;
      fr[(1 + g16) * 16 + slot] = (__bf16)py;
      fr[(2 + g16) * 16 + slot] = (__bf16)1.0f;
    }
    __syncthreads();

    // kp (waves 0-3) and vp (waves 4-7): 16x64 each, K=64
    {
      const int isV = wave >> 2;
      const int rt0 = wave & 3;
      const __bf16* WmF = isV ? sWvF : sWkF;
      const float* bb = isV ? bv : bk;
      for (int ct = 0; ct < 4; ++ct) {
        v8f acc = {0, 0, 0, 0, 0, 0, 0, 0};
        acc = wmma_bf(ld_frag(&sHF[(rt0 * 2 + 0) * 512], lane),
                      ld_frag(&WmF[(0 * 4 + ct) * 512], lane), acc);
        acc = wmma_bf(ld_frag(&sHF[(rt0 * 2 + 1) * 512], lane),
                      ld_frag(&WmF[(1 * 4 + ct) * 512], lane), acc);
        int n = lane & 15, g0 = (lane >> 4) << 3;
        float bcol = bb[ct * 16 + n];
        if (!isV) {  // kp -> A-fragments for the logits GEMM
          __bf16* fr = &sKPF[(rt0 * 2 + (ct >> 1)) * 512];
#pragma unroll
          for (int g = 0; g < 8; ++g)
            fr[frag_off(g0 + g, ((ct & 1) << 4) | n)] = (__bf16)(acc[g] + bcol);
        } else {     // vp -> B-fragments for the pooling GEMM
          __bf16* fr = &sVPF[((rt0 >> 1) * 5 + ct) * 512];
#pragma unroll
          for (int g = 0; g < 8; ++g)
            fr[frag_off(n, ((rt0 & 1) << 4) | (g0 + g))] = (__bf16)(acc[g] + bcol);
        }
      }
    }
    __syncthreads();

    // slot logits: kp[16x64] @ qT[64x16], scale 1/8
    if (wave < 4) {
      const int rt0 = wave;
      v8f acc = {0, 0, 0, 0, 0, 0, 0, 0};
      acc = wmma_bf(ld_frag(&sKPF[(rt0 * 2 + 0) * 512], lane),
                    ld_frag(&sQtF[0], lane), acc);
      acc = wmma_bf(ld_frag(&sKPF[(rt0 * 2 + 1) * 512], lane),
                    ld_frag(&sQtF[512], lane), acc);
      int n = lane & 15, g0 = (lane >> 4) << 3;
#pragma unroll
      for (int g = 0; g < 8; ++g)
        sLog[(rt0 * 16 + g0 + g) * 17 + n] = acc[g] * 0.125f;
    }
    __syncthreads();

    // per-pixel softmax over 16 slots -> m^T A-fragments
    if (tid < 64) {
      const int p = tid, n = n0 + p;
      const int mk = mask[(size_t)b * 2304 + n];
      float l[16], mx = -1e30f;
#pragma unroll
      for (int k = 0; k < 16; ++k) {
        l[k] = mk ? sLog[p * 17 + k] : -10000.0f;
        mx = fmaxf(mx, l[k]);
      }
      float s = 0.0f;
#pragma unroll
      for (int k = 0; k < 16; ++k) { l[k] = __expf(l[k] - mx); s += l[k]; }
      const float inv = mk ? (1.0f / s) : 0.0f;
      int kr = p & 31;
      __bf16* fr = &sMF[(p >> 5) * 512];
      int g16 = (((kr >> 3) & 1) << 4);
      int slot = (kr & 7) + (((kr >> 4) & 1) << 3);
#pragma unroll
      for (int k = 0; k < 16; ++k)
        fr[(k + g16) * 16 + slot] = (__bf16)(l[k] * inv);
    }
    __syncthreads();

    // pooling GEMM: featext[16][80] += m^T[16x64] @ vpext[64x80]
    if (wave < 5) {
      accP = wmma_bf(ld_frag(&sMF[0], lane),
                     ld_frag(&sVPF[(0 * 5 + wave) * 512], lane), accP);
      accP = wmma_bf(ld_frag(&sMF[512], lane),
                     ld_frag(&sVPF[(1 * 5 + wave) * 512], lane), accP);
    }
  }

  if (wave < 5) {
    int n = lane & 15, g0 = (lane >> 4) << 3;
#pragma unroll
    for (int g = 0; g < 8; ++g)
      sOut[(g0 + g) * 80 + wave * 16 + n] = accP[g];
  }
  __syncthreads();
  float* dst = d_part + (size_t)(b * 3 + seg) * 1280;
  for (int i = tid; i < 1280; i += 256) dst[i] = sOut[i];
}

// ===========================================================================
// Stage 2 helper: C[16][ncols] = A[16][kdim] @ W[kdim][ncols] + bias
// ===========================================================================
__device__ void gemm16(const __bf16* sAF, const float* W, int ldw, int kdim,
                       int ncols, const float* bias, float* C, int ldc,
                       __bf16* sBF, int tid) {
  const int lane = tid & 31, wave = tid >> 5;
  for (int c0 = 0; c0 < ncols; c0 += 64) {
    v8f acc = {0, 0, 0, 0, 0, 0, 0, 0};
    for (int k0 = 0; k0 < kdim; k0 += 64) {
      for (int i = tid; i < 64 * 64; i += 256) {
        int r = i >> 6, c = i & 63;
        sBF[((r >> 5) * 4 + (c >> 4)) * 512 + frag_off(c, r & 31)] =
            (__bf16)W[(size_t)(k0 + r) * ldw + c0 + c];
      }
      __syncthreads();
      if (wave < 4) {
        acc = wmma_bf(ld_frag(&sAF[((k0 >> 5) + 0) * 512], lane),
                      ld_frag(&sBF[(0 * 4 + wave) * 512], lane), acc);
        acc = wmma_bf(ld_frag(&sAF[((k0 >> 5) + 1) * 512], lane),
                      ld_frag(&sBF[(1 * 4 + wave) * 512], lane), acc);
      }
      __syncthreads();
    }
    if (wave < 4) {
      int n = lane & 15, g0 = (lane >> 4) << 3;
      int col = c0 + wave * 16 + n;
      float bb = bias ? bias[col] : 0.0f;
#pragma unroll
      for (int g = 0; g < 8; ++g) C[(g0 + g) * ldc + col] = acc[g] + bb;
    }
    __syncthreads();
  }
}

__device__ inline void st_a(__bf16* sAF, int m, int c, float v) {
  sAF[(c >> 5) * 512 + frag_off(m, c & 31)] = (__bf16)v;
}

// ===========================================================================
// Stage 2: per-batch transformer block + class head. Block = batch.
// ===========================================================================
__global__ __launch_bounds__(256) void head_stage(
    const float* __restrict__ d_part, const float* __restrict__ pos_w1,
    const float* __restrict__ pos_b1, const float* __restrict__ pos_w2,
    const float* __restrict__ pos_b2, const float* __restrict__ Wqkv,
    const float* __restrict__ bqkv, const float* __restrict__ Wo,
    const float* __restrict__ bo, const float* __restrict__ ln1_g,
    const float* __restrict__ ln1_b, const float* __restrict__ ffn_w1,
    const float* __restrict__ ffn_b1, const float* __restrict__ ffn_w2,
    const float* __restrict__ ffn_b2, const float* __restrict__ ln2_g,
    const float* __restrict__ ln2_b, const float* __restrict__ cq,
    const float* __restrict__ Wk_cls, const float* __restrict__ bk_cls,
    const float* __restrict__ Wv_cls, const float* __restrict__ bv_cls,
    const float* __restrict__ Wh, const float* __restrict__ bh,
    float* __restrict__ out) {
  const int b = blockIdx.x, tid = threadIdx.x;
  __shared__ v16bf sAF_v[8 * 32];
  __shared__ v16bf sBF_v[8 * 32];
  __shared__ float sX[16 * 64];
  __shared__ float sT[16 * 64];
  __shared__ float sU[16 * 256];
  __shared__ float sCen[32];
  __bf16* sAF = (__bf16*)sAF_v;
  __bf16* sBF = (__bf16*)sBF_v;

  const float* part = d_part + (size_t)b * 3 * 1280;
  for (int i = tid; i < 1280; i += 256)
    sU[i] = part[i] + part[1280 + i] + part[2560 + i];
  __syncthreads();
  for (int i = tid; i < 1024; i += 256) {
    int k = i >> 6, d = i & 63;
    float mass = fmaxf(sU[k * 80 + 66], 1e-6f);
    sX[i] = sU[k * 80 + d] / mass;
  }
  if (tid < 32) {
    int k = tid >> 1, c = tid & 1;
    float mass = fmaxf(sU[k * 80 + 66], 1e-6f);
    sCen[tid] = sU[k * 80 + 64 + c] / mass;
  }
  __syncthreads();

  for (int i = tid; i < 1024; i += 256) {
    int k = i >> 6, d = i & 63;
    float t = sCen[k * 2] * pos_w1[d] + sCen[k * 2 + 1] * pos_w1[64 + d] + pos_b1[d];
    st_a(sAF, k, d, gelu_exact(t));
  }
  __syncthreads();
  gemm16(sAF, pos_w2, 64, 64, 64, pos_b2, sT, 64, sBF, tid);

  for (int i = tid; i < 1024; i += 256) {
    float v = sX[i] + sT[i];
    sX[i] = v;
    st_a(sAF, i >> 6, i & 63, v);
  }
  __syncthreads();

  gemm16(sAF, Wqkv, 192, 64, 192, bqkv, sU, 192, sBF, tid);

  if (tid < 64) {
    const int hh = tid >> 4, row = tid & 15;
    const float* qr = &sU[row * 192 + hh * 16];
    float l[16], mx = -1e30f;
#pragma unroll
    for (int kk = 0; kk < 16; ++kk) {
      const float* kr = &sU[kk * 192 + 64 + hh * 16];
      float s = 0.0f;
      for (int d = 0; d < 16; ++d) s += qr[d] * kr[d];
      l[kk] = s * 0.25f;
      mx = fmaxf(mx, l[kk]);
    }
    float ssum = 0.0f;
#pragma unroll
    for (int kk = 0; kk < 16; ++kk) { l[kk] = __expf(l[kk] - mx); ssum += l[kk]; }
    const float inv = 1.0f / ssum;
    for (int d = 0; d < 16; ++d) {
      float s = 0.0f;
      for (int kk = 0; kk < 16; ++kk) s += l[kk] * sU[kk * 192 + 128 + hh * 16 + d];
      sT[row * 64 + hh * 16 + d] = s * inv;
    }
  }
  __syncthreads();
  for (int i = tid; i < 1024; i += 256) st_a(sAF, i >> 6, i & 63, sT[i]);
  __syncthreads();
  gemm16(sAF, Wo, 64, 64, 64, bo, sT, 64, sBF, tid);

  if (tid < 16) {
    const int r = tid;
    float buf[64], mu = 0.0f;
    for (int d = 0; d < 64; ++d) { buf[d] = sX[r * 64 + d] + sT[r * 64 + d]; mu += buf[d]; }
    mu *= (1.0f / 64.0f);
    float var = 0.0f;
    for (int d = 0; d < 64; ++d) { float t = buf[d] - mu; var += t * t; }
    var *= (1.0f / 64.0f);
    const float rs = rsqrtf(var + 1e-5f);
    for (int d = 0; d < 64; ++d) {
      float v = (buf[d] - mu) * rs * ln1_g[d] + ln1_b[d];
      sX[r * 64 + d] = v;
      st_a(sAF, r, d, v);
    }
  }
  __syncthreads();

  gemm16(sAF, ffn_w1, 256, 64, 256, ffn_b1, sU, 256, sBF, tid);
  for (int i = tid; i < 16 * 256; i += 256)
    st_a(sAF, i >> 8, i & 255, gelu_exact(sU[i]));
  __syncthreads();
  gemm16(sAF, ffn_w2, 64, 256, 64, ffn_b2, sT, 64, sBF, tid);

  if (tid < 16) {
    const int r = tid;
    float buf[64], mu = 0.0f;
    for (int d = 0; d < 64; ++d) { buf[d] = sX[r * 64 + d] + sT[r * 64 + d]; mu += buf[d]; }
    mu *= (1.0f / 64.0f);
    float var = 0.0f;
    for (int d = 0; d < 64; ++d) { float t = buf[d] - mu; var += t * t; }
    var *= (1.0f / 64.0f);
    const float rs = rsqrtf(var + 1e-5f);
    for (int d = 0; d < 64; ++d) {
      float v = (buf[d] - mu) * rs * ln2_g[d] + ln2_b[d];
      sX[r * 64 + d] = v;
      st_a(sAF, r, d, v);
    }
  }
  __syncthreads();

  gemm16(sAF, Wk_cls, 64, 64, 64, bk_cls, &sU[0], 64, sBF, tid);
  gemm16(sAF, Wv_cls, 64, 64, 64, bv_cls, &sU[1024], 64, sBF, tid);

  if (tid < 7) {
    const int c = tid;
    float l[16], mx = -1e30f;
#pragma unroll
    for (int k = 0; k < 16; ++k) {
      float s = 0.0f;
      for (int d = 0; d < 64; ++d) s += cq[c * 64 + d] * sU[k * 64 + d];
      l[k] = s * 0.125f;
      mx = fmaxf(mx, l[k]);
    }
    float ss = 0.0f;
#pragma unroll
    for (int k = 0; k < 16; ++k) { l[k] = __expf(l[k] - mx); ss += l[k]; }
    const float inv = 1.0f / ss;
    float acc = 0.0f;
    for (int d = 0; d < 64; ++d) {
      float r = 0.0f;
      for (int k = 0; k < 16; ++k) r += l[k] * sU[1024 + k * 64 + d];
      acc += r * inv * Wh[d];
    }
    out[b * 7 + c] = acc + bh[0];
  }
}

// ===========================================================================
extern "C" void kernel_launch(void* const* d_in, const int* in_sizes, int n_in,
                              void* d_out, int out_size, void* d_ws,
                              size_t ws_size, hipStream_t stream) {
  const float* h      = (const float*)d_in[0];
  const int*   mask   = (const int*)d_in[1];
  const float* pq     = (const float*)d_in[2];
  const float* Wk_pix = (const float*)d_in[3];
  const float* bk_pix = (const float*)d_in[4];
  const float* Wv_pix = (const float*)d_in[5];
  const float* bv_pix = (const float*)d_in[6];
  const float* pos_w1 = (const float*)d_in[7];
  const float* pos_b1 = (const float*)d_in[8];
  const float* pos_w2 = (const float*)d_in[9];
  const float* pos_b2 = (const float*)d_in[10];
  const float* Wqkv   = (const float*)d_in[11];
  const float* bqkv   = (const float*)d_in[12];
  const float* Wo     = (const float*)d_in[13];
  const float* bo     = (const float*)d_in[14];
  const float* ln1_g  = (const float*)d_in[15];
  const float* ln1_b  = (const float*)d_in[16];
  const float* ffn_w1 = (const float*)d_in[17];
  const float* ffn_b1 = (const float*)d_in[18];
  const float* ffn_w2 = (const float*)d_in[19];
  const float* ffn_b2 = (const float*)d_in[20];
  const float* ln2_g  = (const float*)d_in[21];
  const float* ln2_b  = (const float*)d_in[22];
  const float* cq     = (const float*)d_in[23];
  const float* Wk_cls = (const float*)d_in[24];
  const float* bk_cls = (const float*)d_in[25];
  const float* Wv_cls = (const float*)d_in[26];
  const float* bv_cls = (const float*)d_in[27];
  const float* Wh     = (const float*)d_in[28];
  const float* bh     = (const float*)d_in[29];

  float* d_part = (float*)d_ws;  // [B][3][16][80] f32 = 3.93 MB

  dim3 g1(256, 3);
  pixel_stage<<<g1, 256, 0, stream>>>(h, mask, pq, Wk_pix, bk_pix, Wv_pix,
                                      bv_pix, d_part);
  head_stage<<<256, 256, 0, stream>>>(
      d_part, pos_w1, pos_b1, pos_w2, pos_b2, Wqkv, bqkv, Wo, bo, ln1_g, ln1_b,
      ffn_w1, ffn_b1, ffn_w2, ffn_b2, ln2_g, ln2_b, cq, Wk_cls, bk_cls, Wv_cls,
      bv_cls, Wh, bh, (float*)d_out);
}
